// Block_17987323036347
// MI455X (gfx1250) — matrix operand order repaired and checked by reading, not compiled, not run on hardware
//
#include <hip/hip_runtime.h>

// ---------------------------------------------------------------------------
// GPT block for MI455X (gfx1250): wave32, WMMA f32_16x16x32_f16,
// double-buffered async global->LDS staging (ASYNCcnt), pre-transposed
// weights/V so the hot loops have zero VGPR-bounce transposes.
// B=4, T=2048, C=1024, H=16, HD=64.
// ---------------------------------------------------------------------------

typedef _Float16 h16;
typedef __attribute__((ext_vector_type(8)))  _Float16 v8h;
typedef __attribute__((ext_vector_type(16))) _Float16 v16h;
typedef __attribute__((ext_vector_type(8)))  float    v8f;

#define GB   4
#define GT   2048
#define GC   1024
#define GH   16
#define GHD  64
#define GM   (GB * GT)      // 8192 rows
#define G3C  (3 * GC)       // 3072

// ---- CDNA5 async global->LDS (16B per lane), tracked by ASYNCcnt ----------
#define USE_ASYNC 1

static __device__ __forceinline__ void async_ld_b128(h16* lds, const h16* g) {
#if USE_ASYNC
  unsigned l32 = (unsigned)(uintptr_t)lds;           // low 32 bits = LDS offset
  unsigned long long g64 = (unsigned long long)(uintptr_t)g;
  asm volatile("global_load_async_to_lds_b128 %0, %1, off"
               :: "v"(l32), "v"(g64) : "memory");
#else
  *(v8h*)lds = *(const v8h*)g;
#endif
}
static __device__ __forceinline__ void wait_async0() {
#if USE_ASYNC
  asm volatile("s_wait_asynccnt 0x0" ::: "memory");
#endif
}
// allow the newest 4 async ops (next tile) to stay in flight
static __device__ __forceinline__ void wait_async4() {
#if USE_ASYNC
  asm volatile("s_wait_asynccnt 0x4" ::: "memory");
#endif
}

static __device__ __forceinline__ v16h cat8(v8h lo, v8h hi) {
  v16h r;
#pragma unroll
  for (int j = 0; j < 8; ++j) { r[j] = lo[j]; r[j + 8] = hi[j]; }
  return r;
}

static __device__ __forceinline__ v8f wmma_f16(v16h a, v16h b, v8f c) {
  return __builtin_amdgcn_wmma_f32_16x16x32_f16(false, a, false, b,
                                                (short)0, c, false, false);
}

// tanh-gelu via the identity 0.5*x*(1+tanh(u)) == x*sigmoid(2u):
// one v_exp_f32 instead of a libm tanhf call.
static __device__ __forceinline__ float gelu_tanh_f(float x) {
  float u2 = 1.5957691216057308f * (x + 0.044715f * x * x * x);  // 2*sqrt(2/pi)*(...)
  return x / (1.0f + __expf(-u2));
}

// ---------------------------------------------------------------------------
// Weight convert + transpose: in[K,N] f32 -> out[N,K] f16 (32x32 LDS tiles)
// ---------------------------------------------------------------------------
__global__ __launch_bounds__(256) void cvt_transpose(const float* __restrict__ in,
                                                     h16* __restrict__ out,
                                                     int K, int N) {
  __shared__ h16 tl[32][40];
  const int n0 = blockIdx.x * 32, k0 = blockIdx.y * 32;
  const int tx = threadIdx.x, ty = threadIdx.y;  // (32, 8)
#pragma unroll
  for (int r = 0; r < 4; ++r) {
    int k = ty + r * 8;
    tl[k][tx] = (h16)in[(size_t)(k0 + k) * N + n0 + tx];
  }
  __syncthreads();
#pragma unroll
  for (int r = 0; r < 4; ++r) {
    int n = ty + r * 8;
    out[(size_t)(n0 + n) * K + k0 + tx] = tl[tx][n];
  }
}

// ---------------------------------------------------------------------------
// V transpose: qkv v-part -> vt[B*H][64][T]  (once, instead of per q-tile)
// grid (T/32, B*H), 256 thr
// ---------------------------------------------------------------------------
__global__ __launch_bounds__(256) void v_transpose(const h16* __restrict__ qkv,
                                                   h16* __restrict__ vt) {
  __shared__ h16 tl[64][40];
  const int bh = blockIdx.y, bb = bh >> 4, hh = bh & 15;
  const int t0 = blockIdx.x * 32;
  const int tid = threadIdx.x;
  {
    int linear = tid * 8;  // over 32 t x 64 d
    int tloc = linear >> 6, d = linear & 63;
    v8h vv = *(const v8h*)(qkv + (size_t)(bb * GT + t0 + tloc) * G3C +
                           2 * GC + hh * GHD + d);
#pragma unroll
    for (int j = 0; j < 8; ++j) tl[d + j][tloc] = vv[j];
  }
  __syncthreads();
  {
    int linear = tid * 8;  // over 64 d x 32 t
    int d = linear >> 5, tloc = linear & 31;
    *(v8h*)(vt + ((size_t)bh * GHD + d) * GT + t0 + tloc) =
        *(const v8h*)&tl[d][tloc];
  }
}

// ---------------------------------------------------------------------------
// LayerNorm over C=1024, one block per row, output f16
// ---------------------------------------------------------------------------
__global__ __launch_bounds__(256) void ln_kernel(const float* __restrict__ x,
                                                 const float* __restrict__ wln,
                                                 const float* __restrict__ bln,
                                                 h16* __restrict__ out) {
  int row = blockIdx.x;
  int t = threadIdx.x;
  __shared__ float rs[256], rss[256];
  const float* xr = x + (size_t)row * GC;
  float lv[4], s = 0.f, ss = 0.f;
#pragma unroll
  for (int i = 0; i < 4; ++i) {
    lv[i] = xr[t + i * 256];
    s += lv[i];
    ss += lv[i] * lv[i];
  }
  rs[t] = s; rss[t] = ss;
  __syncthreads();
  for (int off = 128; off > 0; off >>= 1) {
    if (t < off) { rs[t] += rs[t + off]; rss[t] += rss[t + off]; }
    __syncthreads();
  }
  float mu = rs[0] * (1.0f / GC);
  float var = rss[0] * (1.0f / GC) - mu * mu;
  float inv = rsqrtf(var + 1e-5f);
#pragma unroll
  for (int i = 0; i < 4; ++i) {
    int c = t + i * 256;
    out[(size_t)row * GC + c] = (h16)((lv[i] - mu) * inv * wln[c] + bln[c]);
  }
}

// ---------------------------------------------------------------------------
// Tiled WMMA GEMM: out[M,N] = A[M,K](f16) @ Wt[N,K](f16, pre-transposed) + bias
// Block: 256 thr (8 waves), 128x128 tile; wave = 64x32 (4x2 wmma accums).
// Double-buffered pure-async staging: 4 async b128 per thread per K-step.
// MODE 0: f16 out. MODE 1: f16 gelu out. MODE 2: f32 out = res + acc + bias.
// ---------------------------------------------------------------------------
#define GP 40  // LDS row pitch (halves): 20 dwords -> lane->bank permutation

template <int MODE>
__global__ __launch_bounds__(256) void gemm_kernel(
    const h16* __restrict__ A, const h16* __restrict__ Wt,
    const float* __restrict__ bias, h16* __restrict__ out_h,
    const float* __restrict__ res, float* __restrict__ out_f,
    int M, int N, int K) {
  __shared__ h16 lA[2][128 * GP];
  __shared__ h16 lB[2][128 * GP];  // Wt tile: lB[n][k]

  const int m0 = blockIdx.y * 128;
  const int n0 = blockIdx.x * 128;
  const int tid = threadIdx.x;
  const int wid = tid >> 5;
  const int lane = tid & 31;
  const int lm = lane & 15, lh = lane >> 4;
  const int wm = wid >> 2, wn = wid & 3;  // wave grid 2(m) x 4(n)

  // staging geometry: 128 rows x 32 halves per tile, 16 halves per thread
  const int sr = tid >> 1;            // 0..127
  const int sc = (tid & 1) * 16;      // 0 or 16

  v8f acc[4][2];
  {
    v8f z = {};
#pragma unroll
    for (int i = 0; i < 4; ++i) { acc[i][0] = z; acc[i][1] = z; }
  }

  auto stage = [&](int k0, int buf) {
    const h16* ga = A + (size_t)(m0 + sr) * K + k0 + sc;
    async_ld_b128(&lA[buf][sr * GP + sc], ga);
    async_ld_b128(&lA[buf][sr * GP + sc + 8], ga + 8);
    const h16* gb = Wt + (size_t)(n0 + sr) * K + k0 + sc;
    async_ld_b128(&lB[buf][sr * GP + sc], gb);
    async_ld_b128(&lB[buf][sr * GP + sc + 8], gb + 8);
  };

  const int nsteps = K >> 5;
  stage(0, 0);
  for (int i = 0; i < nsteps; ++i) {
    const int buf = i & 1;
    if (i + 1 < nsteps) {
      stage((i + 1) << 5, buf ^ 1);
      wait_async4();              // tile i landed; tile i+1 stays in flight
    } else {
      wait_async0();
    }
    __syncthreads();

    v16h afrag[4], bfrag[2];
#pragma unroll
    for (int sm = 0; sm < 4; ++sm) {
      int row = wm * 64 + sm * 16 + lm;
      v8h lo = *(const v8h*)&lA[buf][row * GP + lh * 16];
      v8h hi = *(const v8h*)&lA[buf][row * GP + lh * 16 + 8];
      afrag[sm] = cat8(lo, hi);
    }
#pragma unroll
    for (int sn = 0; sn < 2; ++sn) {
      int col = wn * 32 + sn * 16 + lm;
      v8h lo = *(const v8h*)&lB[buf][col * GP + lh * 16];
      v8h hi = *(const v8h*)&lB[buf][col * GP + lh * 16 + 8];
      bfrag[sn] = cat8(lo, hi);
    }
#pragma unroll
    for (int sm = 0; sm < 4; ++sm)
#pragma unroll
      for (int sn = 0; sn < 2; ++sn)
        acc[sm][sn] = wmma_f16(afrag[sm], bfrag[sn], acc[sm][sn]);

    __syncthreads();  // all waves done reading buf before it is re-staged
  }

  // epilogue
#pragma unroll
  for (int sm = 0; sm < 4; ++sm) {
#pragma unroll
    for (int sn = 0; sn < 2; ++sn) {
      int gn = n0 + wn * 32 + sn * 16 + lm;
      float bv = bias[gn];
#pragma unroll
      for (int r = 0; r < 8; ++r) {
        int gm = m0 + wm * 64 + sm * 16 + r + lh * 8;
        float v = acc[sm][sn][r] + bv;
        if (MODE == 0) {
          out_h[(size_t)gm * N + gn] = (h16)v;
        } else if (MODE == 1) {
          out_h[(size_t)gm * N + gn] = (h16)gelu_tanh_f(v);
        } else {
          out_f[(size_t)gm * N + gn] = res[(size_t)gm * N + gn] + v;
        }
      }
    }
  }
}

// ---------------------------------------------------------------------------
// Flash attention (causal). Grid: (T/64, B*H). Block: 128 thr = 4 waves.
// Wave w handles 16 query rows; key tiles of 32, double-buffered async
// staging of K (row-major from qkv) and V^T (from pre-transposed vt).
// ---------------------------------------------------------------------------
#define KPITCH 72
#define VPITCH 40

__global__ __launch_bounds__(128) void attn_kernel(const h16* __restrict__ qkv,
                                                   const h16* __restrict__ vt,
                                                   h16* __restrict__ y) {
  __shared__ h16 lK[2][32 * KPITCH];
  __shared__ h16 lV[2][64 * VPITCH];     // transposed V: lV[d][k]
  __shared__ float sS[4][16 * 32];
  __shared__ h16 sP[4][16 * VPITCH];
  __shared__ float sM[4][16], sL[4][16], sScale[4][16];

  const int qt = blockIdx.x;
  const int bh = blockIdx.y;
  const int bb = bh >> 4, hh = bh & 15;
  const int tid = threadIdx.x;
  const int wid = tid >> 5;
  const int lane = tid & 31;
  const int lm = lane & 15, lh = lane >> 4;
  const int qrow0 = qt * 64 + wid * 16;

  // staging geometry (128 threads, 16 halves each per tile)
  const int kr = tid >> 2, kc = (tid & 3) * 16;  // K tile 32x64
  const int vr = tid >> 1, vc = (tid & 1) * 16;  // V tile 64x32

  auto stage = [&](int kt, int buf) {
    const h16* gk =
        qkv + (size_t)(bb * GT + kt * 32 + kr) * G3C + GC + hh * GHD + kc;
    async_ld_b128(&lK[buf][kr * KPITCH + kc], gk);
    async_ld_b128(&lK[buf][kr * KPITCH + kc + 8], gk + 8);
    const h16* gv = vt + ((size_t)bh * GHD + vr) * GT + kt * 32 + vc;
    async_ld_b128(&lV[buf][vr * VPITCH + vc], gv);
    async_ld_b128(&lV[buf][vr * VPITCH + vc + 8], gv + 8);
  };

  // load Q fragments (scaled by 1/sqrt(64))
  v16h qf[2];
  {
    const h16* qp = qkv + (size_t)(bb * GT + qrow0 + lm) * G3C + hh * GHD;
#pragma unroll
    for (int f = 0; f < 2; ++f) {
      int d0 = f * 32 + lh * 16;
      v8h lo = *(const v8h*)(qp + d0);
      v8h hi = *(const v8h*)(qp + d0 + 8);
      qf[f] = cat8(lo, hi);
#pragma unroll
      for (int j = 0; j < 16; ++j) qf[f][j] = (h16)((float)qf[f][j] * 0.125f);
    }
  }

  if (lane < 16) {
    sM[wid][lane] = -1e30f;
    sL[wid][lane] = 0.0f;
  }
  v8f o[4];
  {
    v8f z = {};
    o[0] = z; o[1] = z; o[2] = z; o[3] = z;
  }

  const int nkt = (qt + 1) * 2;  // key tiles of 32 needed (causal)
  stage(0, 0);
  for (int kt = 0; kt < nkt; ++kt) {
    const int buf = kt & 1;
    if (kt + 1 < nkt) {
      stage(kt + 1, buf ^ 1);
      wait_async4();
    } else {
      wait_async0();
    }
    __syncthreads();

    // S = Q @ K^T for two 16-key subtiles
#pragma unroll
    for (int s = 0; s < 2; ++s) {
      int col = s * 16 + lm;
      v16h kf0 = cat8(*(const v8h*)&lK[buf][col * KPITCH + lh * 16],
                      *(const v8h*)&lK[buf][col * KPITCH + lh * 16 + 8]);
      v16h kf1 = cat8(*(const v8h*)&lK[buf][col * KPITCH + 32 + lh * 16],
                      *(const v8h*)&lK[buf][col * KPITCH + 32 + lh * 16 + 8]);
      v8f sacc = {};
      sacc = wmma_f16(qf[0], kf0, sacc);
      sacc = wmma_f16(qf[1], kf1, sacc);
#pragma unroll
      for (int r = 0; r < 8; ++r)
        sS[wid][(r + lh * 8) * 32 + s * 16 + lm] = sacc[r];
    }
    __syncthreads();

    // online softmax: 2 lanes per row, 16 cols each
    {
      int row = lane >> 1;
      int cbase = (lane & 1) * 16;
      int gq = qt * 64 + wid * 16 + row;
      float sv[16];
      float mloc = -1e30f;
#pragma unroll
      for (int j = 0; j < 16; ++j) {
        int gk = kt * 32 + cbase + j;
        float v = sS[wid][row * 32 + cbase + j];
        if (gk > gq) v = -1e30f;
        sv[j] = v;
        mloc = fmaxf(mloc, v);
      }
      mloc = fmaxf(mloc, __shfl_xor(mloc, 1, 32));
      float mold = sM[wid][row];
      float mnew = fmaxf(mold, mloc);
      float lsum = 0.0f;
#pragma unroll
      for (int j = 0; j < 16; ++j) {
        float p = __expf(sv[j] - mnew);
        lsum += p;
        sP[wid][row * VPITCH + cbase + j] = (h16)p;
      }
      lsum += __shfl_xor(lsum, 1, 32);
      float sc = __expf(mold - mnew);
      if ((lane & 1) == 0) {
        sM[wid][row] = mnew;
        sL[wid][row] = sL[wid][row] * sc + lsum;
        sScale[wid][row] = sc;
      }
    }
    __syncthreads();

    // rescale O accumulators by exp(m_old - m_new) per row
    float scl[8];
#pragma unroll
    for (int r = 0; r < 8; ++r) scl[r] = sScale[wid][r + lh * 8];
#pragma unroll
    for (int ds = 0; ds < 4; ++ds)
#pragma unroll
      for (int r = 0; r < 8; ++r) o[ds][r] *= scl[r];

    // P fragment (A layout, K-dim = 32 keys)
    v16h pf = cat8(*(const v8h*)&sP[wid][lm * VPITCH + lh * 16],
                   *(const v8h*)&sP[wid][lm * VPITCH + lh * 16 + 8]);
    // O += P @ V  (4 d-subtiles of 16)
#pragma unroll
    for (int ds = 0; ds < 4; ++ds) {
      v16h vf = cat8(*(const v8h*)&lV[buf][(ds * 16 + lm) * VPITCH + lh * 16],
                     *(const v8h*)&lV[buf][(ds * 16 + lm) * VPITCH + lh * 16 + 8]);
      o[ds] = wmma_f16(pf, vf, o[ds]);
    }
    __syncthreads();  // buf free for re-staging; sS/sP free for next tile
  }

  // epilogue: O /= l, store y f16 [B*T, C]
  float linv[8];
#pragma unroll
  for (int r = 0; r < 8; ++r) linv[r] = 1.0f / sL[wid][r + lh * 8];
#pragma unroll
  for (int ds = 0; ds < 4; ++ds) {
#pragma unroll
    for (int r = 0; r < 8; ++r) {
      int gm = bb * GT + qt * 64 + wid * 16 + r + lh * 8;
      int gc = hh * GHD + ds * 16 + lm;
      y[(size_t)gm * GC + gc] = (h16)(o[ds][r] * linv[r]);
    }
  }
}

// ---------------------------------------------------------------------------
// Host launcher
// ---------------------------------------------------------------------------
static size_t align256(size_t x) { return (x + 255) & ~(size_t)255; }

extern "C" void kernel_launch(void* const* d_in, const int* in_sizes, int n_in,
                              void* d_out, int out_size, void* d_ws,
                              size_t ws_size, hipStream_t stream) {
  (void)in_sizes; (void)n_in; (void)out_size; (void)ws_size;

  const float* x         = (const float*)d_in[0];
  const float* ln1_w     = (const float*)d_in[1];
  const float* ln1_b     = (const float*)d_in[2];
  const float* attn_w    = (const float*)d_in[3];
  const float* attn_b    = (const float*)d_in[4];
  const float* proj_w    = (const float*)d_in[5];
  const float* proj_b    = (const float*)d_in[6];
  const float* ln2_w     = (const float*)d_in[7];
  const float* ln2_b     = (const float*)d_in[8];
  const float* fc_w      = (const float*)d_in[9];
  const float* fc_b      = (const float*)d_in[10];
  const float* fc_proj_w = (const float*)d_in[11];
  const float* fc_proj_b = (const float*)d_in[12];
  float* out = (float*)d_out;

  // workspace carve-up
  char* ws = (char*)d_ws;
  size_t off = 0;
  h16* h1    = (h16*)(ws + off); off = align256(off + (size_t)GM * GC * 2);
  h16* qkv_h = (h16*)(ws + off); off = align256(off + (size_t)GM * G3C * 2);
  h16* y_h   = (h16*)(ws + off); off = align256(off + (size_t)GM * GC * 2);
  float* x2  = (float*)(ws + off); off = align256(off + (size_t)GM * GC * 4);
  h16* h2    = (h16*)(ws + off); off = align256(off + (size_t)GM * GC * 2);
  h16* hfc   = (h16*)(ws + off); off = align256(off + (size_t)GM * 4 * GC * 2);
  h16* vt    = (h16*)(ws + off); off = align256(off + (size_t)GB * GH * GHD * GT * 2);
  h16* awT   = (h16*)(ws + off); off = align256(off + (size_t)GC * G3C * 2);
  h16* pwT   = (h16*)(ws + off); off = align256(off + (size_t)GC * GC * 2);
  h16* fwT   = (h16*)(ws + off); off = align256(off + (size_t)GC * 4 * GC * 2);
  h16* fpwT  = (h16*)(ws + off); off = align256(off + (size_t)4 * GC * GC * 2);

  dim3 tblk(32, 8);
  // 1. convert + transpose weights to f16 [N,K]
  cvt_transpose<<<dim3(G3C / 32, GC / 32), tblk, 0, stream>>>(attn_w, awT, GC, G3C);
  cvt_transpose<<<dim3(GC / 32, GC / 32), tblk, 0, stream>>>(proj_w, pwT, GC, GC);
  cvt_transpose<<<dim3(4 * GC / 32, GC / 32), tblk, 0, stream>>>(fc_w, fwT, GC, 4 * GC);
  cvt_transpose<<<dim3(GC / 32, 4 * GC / 32), tblk, 0, stream>>>(fc_proj_w, fpwT, 4 * GC, GC);

  // 2. h1 = LN1(x)  -> f16
  ln_kernel<<<GM, 256, 0, stream>>>(x, ln1_w, ln1_b, h1);

  // 3. qkv = h1 @ attn_w + attn_b  -> f16
  gemm_kernel<0><<<dim3(G3C / 128, GM / 128), 256, 0, stream>>>(
      h1, awT, attn_b, qkv_h, nullptr, nullptr, GM, G3C, GC);

  // 3b. vt = V^T per (b,h)
  v_transpose<<<dim3(GT / 32, GB * GH), 256, 0, stream>>>(qkv_h, vt);

  // 4. y = flash-attention(qkv, vt)  -> f16
  attn_kernel<<<dim3(GT / 64, GB * GH), 128, 0, stream>>>(qkv_h, vt, y_h);

  // 5. x2 = x + y @ proj_w + proj_b  -> f32
  gemm_kernel<2><<<dim3(GC / 128, GM / 128), 256, 0, stream>>>(
      y_h, pwT, proj_b, nullptr, x, x2, GM, GC, GC);

  // 6. h2 = LN2(x2) -> f16
  ln_kernel<<<GM, 256, 0, stream>>>(x2, ln2_w, ln2_b, h2);

  // 7. hfc = gelu(h2 @ fc_w + fc_b) -> f16
  gemm_kernel<1><<<dim3(4 * GC / 128, GM / 128), 256, 0, stream>>>(
      h2, fwT, fc_b, hfc, nullptr, nullptr, GM, 4 * GC, GC);

  // 8. out = x2 + hfc @ fc_proj_w + fc_proj_b -> f32
  gemm_kernel<2><<<dim3(GC / 128, GM / 128), 256, 0, stream>>>(
      hfc, fpwT, fc_proj_b, nullptr, x2, out, GM, GC, 4 * GC);
}
